// GlobalLocalChannelAttention_12481174962662
// MI455X (gfx1250) — matrix operand order repaired and checked by reading, not compiled, not run on hardware
//
#include <hip/hip_runtime.h>

typedef __attribute__((ext_vector_type(16))) __bf16 v16bf;
typedef __attribute__((ext_vector_type(8)))  float  v8f;

union FragB16 {
  v16bf v;
  uint4 u[2];
};

#define B_    8
#define HW_   4096
#define C_    1024
#define DH_   256
#define SCALE_ (0.03125f)                        /* 1/sqrt(1024) */
#define KLOG_  (SCALE_ * 1.4426950408889634f)    /* scale * log2(e) */

// ---------------------------------------------------------------------------
// Kernel 1: P = softmax(scale * (Wq.q) (Wk.k)^T) per branch, per (b,h).
// Block = 256 threads (8 waves), tile = 64 rows x 256 cols, K = 4096.
// ---------------------------------------------------------------------------
__global__ __launch_bounds__(256) void k_qk_softmax(
    const float* __restrict__ q, const float* __restrict__ kk,
    const float* __restrict__ wq_g, const float* __restrict__ bq_g,
    const float* __restrict__ wk_g, const float* __restrict__ bk_g,
    const float* __restrict__ wq_l, const float* __restrict__ bq_l,
    const float* __restrict__ wk_l, const float* __restrict__ bk_l,
    __bf16* __restrict__ p_ws)
{
  __shared__ __bf16 lds_a[64 * 40];    // [chan][k], stride 40 bf16 (80B, 16B-aligned rows)
  __shared__ __bf16 lds_b[256 * 40];   // [chan][k]
  __shared__ float  lds_s[64 * 257];   // S staging, stride 257 to avoid bank conflicts
  __shared__ float  rmax[64], rinv[64];

  const int bx = blockIdx.x;
  const int mt = bx & 3;               // M tile (64 q-channels)
  const int bh = (bx >> 2) & 31;       // b*4 + h
  const int br = bx >> 7;              // branch: 0 global, 1 local
  const int b  = bh >> 2;
  const int h  = bh & 3;

  const float* Wq = br ? wq_l : wq_g;
  const float* Bq = br ? bq_l : bq_g;
  const float* Wk = br ? wk_l : wk_g;
  const float* Bk = br ? bk_l : bk_g;

  const int tid  = threadIdx.x;
  const int lane = tid & 31;
  const int hl   = lane >> 4;          // wave half
  const int ln   = lane & 15;
  const int w    = tid >> 5;
  const int wm   = w >> 2;             // 0..1 (32-row slab)
  const int wn   = w & 3;              // 0..3 (64-col slab)

  // A staging mapping: one token (tA), 8 channels (cA0..+7)
  const int tA  = tid >> 3;            // 0..31
  const int cA0 = (tid & 7) * 8;       // 0..56
  const int gA  = h * DH_ + mt * 64 + cA0;
  float wa[8], ba[8];
#pragma unroll
  for (int j = 0; j < 8; ++j) { wa[j] = Wq[gA + j]; ba[j] = Bq[gA + j]; }

  // B staging mapping: 4 tokens (tB0..+3), 8 channels (cB0..+7)
  const int cB0 = (tid & 31) * 8;      // 0..248
  const int tB0 = (tid >> 5) * 4;      // 0..28
  const int gB  = h * DH_ + cB0;
  float wb[8], bb[8];
#pragma unroll
  for (int j = 0; j < 8; ++j) { wb[j] = Wk[gB + j]; bb[j] = Bk[gB + j]; }

  v8f acc[2][4] = {};

  for (int kc = 0; kc < HW_; kc += 32) {
    // --- stage A tile (64 ch x 32 tok), fused affine, fp32 -> bf16, transpose
    {
      const float* qp = q + ((size_t)b * HW_ + kc + tA) * C_ + gA;
      float4 x0 = *(const float4*)(qp);
      float4 x1 = *(const float4*)(qp + 4);
      float xv[8] = {x0.x, x0.y, x0.z, x0.w, x1.x, x1.y, x1.z, x1.w};
#pragma unroll
      for (int j = 0; j < 8; ++j)
        lds_a[(cA0 + j) * 40 + tA] = (__bf16)(xv[j] * wa[j] + ba[j]);
    }
    // --- stage B tile (256 ch x 32 tok)
#pragma unroll
    for (int j2 = 0; j2 < 4; ++j2) {
      const int t = tB0 + j2;
      const float* kp = kk + ((size_t)b * HW_ + kc + t) * C_ + gB;
      float4 x0 = *(const float4*)(kp);
      float4 x1 = *(const float4*)(kp + 4);
      float xv[8] = {x0.x, x0.y, x0.z, x0.w, x1.x, x1.y, x1.z, x1.w};
#pragma unroll
      for (int j = 0; j < 8; ++j)
        lds_b[(cB0 + j) * 40 + t] = (__bf16)(xv[j] * wb[j] + bb[j]);
    }
    __syncthreads();

    // --- fragment loads (ISA A layout: K = hl*8+{0..7} and 16+hl*8+{0..7})
    FragB16 af[2];
#pragma unroll
    for (int mi = 0; mi < 2; ++mi) {
      const int row = wm * 32 + mi * 16 + ln;
      af[mi].u[0] = *(const uint4*)&lds_a[row * 40 + hl * 8];
      af[mi].u[1] = *(const uint4*)&lds_a[row * 40 + 16 + hl * 8];
    }
    // --- B layout: K = hl*8+{0..7} and 16+hl*8+{0..7}, col = ln
    FragB16 bfr[4];
#pragma unroll
    for (int ni = 0; ni < 4; ++ni) {
      const int col = wn * 64 + ni * 16 + ln;
      bfr[ni].u[0] = *(const uint4*)&lds_b[col * 40 + hl * 8];
      bfr[ni].u[1] = *(const uint4*)&lds_b[col * 40 + 16 + hl * 8];
    }
#pragma unroll
    for (int mi = 0; mi < 2; ++mi)
#pragma unroll
      for (int ni = 0; ni < 4; ++ni)
        acc[mi][ni] = __builtin_amdgcn_wmma_f32_16x16x32_bf16(
            false, af[mi].v, false, bfr[ni].v, (short)0, acc[mi][ni],
            false, false);
    __syncthreads();
  }

  // --- dump S tile to LDS (C/D layout: row = i + 8*hl, col = ln)
#pragma unroll
  for (int mi = 0; mi < 2; ++mi)
#pragma unroll
    for (int ni = 0; ni < 4; ++ni)
#pragma unroll
      for (int i = 0; i < 8; ++i) {
        const int row = wm * 32 + mi * 16 + i + 8 * hl;
        const int col = wn * 64 + ni * 16 + ln;
        lds_s[row * 257 + col] = acc[mi][ni][i];
      }
  __syncthreads();

  // --- row-wise softmax statistics (64 rows, 256 wide)
  if (tid < 64) {
    float m = -3.4e38f;
    for (int c = 0; c < 256; ++c) m = fmaxf(m, lds_s[tid * 257 + c]);
    float s = 0.f;
    for (int c = 0; c < 256; ++c) s += exp2f((lds_s[tid * 257 + c] - m) * KLOG_);
    rmax[tid] = m;
    rinv[tid] = 1.0f / s;
  }
  __syncthreads();

  // --- write P (bf16) to workspace: [br][bh][256 q-ch][256 k-ch]
  {
    const int r  = tid >> 2;
    const int c0 = (tid & 3) * 64;
    const float m   = rmax[r];
    const float inv = rinv[r];
    const size_t base =
        (((size_t)br * 32 + bh) * 256 + (mt * 64 + r)) * 256 + c0;
#pragma unroll 4
    for (int j = 0; j < 64; ++j) {
      const float p = exp2f((lds_s[r * 257 + c0 + j] - m) * KLOG_) * inv;
      p_ws[base + j] = (__bf16)p;
    }
  }
}

// ---------------------------------------------------------------------------
// Kernel 2: out = Wp.(Pg Vg + Pl Vl) + bp.   Block tile: 256 ch x 64 tokens.
// v read once; both branch affines produced from the same load.
// ---------------------------------------------------------------------------
__global__ __launch_bounds__(256) void k_pv_out(
    const float* __restrict__ v, const __bf16* __restrict__ p_ws,
    const float* __restrict__ wv_g, const float* __restrict__ bv_g,
    const float* __restrict__ wv_l, const float* __restrict__ bv_l,
    const float* __restrict__ wp, const float* __restrict__ bp,
    float* __restrict__ out)
{
  __shared__ __bf16 lds_vg[64 * 40];   // [token][chan-chunk], stride 40 bf16
  __shared__ __bf16 lds_vl[64 * 40];
  __shared__ float  lds_o[64 * 257];   // [token][chan] output staging

  const int bx = blockIdx.x;
  const int nt = bx & 63;              // token tile
  const int bh = bx >> 6;
  const int b  = bh >> 2;
  const int h  = bh & 3;
  const int tbase = nt * 64;

  const int tid  = threadIdx.x;
  const int lane = tid & 31;
  const int hl   = lane >> 4;
  const int ln   = lane & 15;
  const int w    = tid >> 5;
  const int wm   = w >> 1;             // 0..3 (64-channel slab)
  const int wn   = w & 1;              // 0..1 (32-token slab)

  const int tV  = tid & 63;
  const int cV0 = (tid >> 6) * 8;

  v8f acc[4][2] = {};

  for (int kc = 0; kc < 256; kc += 32) {
    // --- stage V chunk once, emit both branch-affine bf16 tiles
    {
      const int gc = h * DH_ + kc + cV0;
      const float* vp = v + ((size_t)b * HW_ + tbase + tV) * C_ + gc;
      float4 x0 = *(const float4*)(vp);
      float4 x1 = *(const float4*)(vp + 4);
      float xv[8] = {x0.x, x0.y, x0.z, x0.w, x1.x, x1.y, x1.z, x1.w};
#pragma unroll
      for (int j = 0; j < 8; ++j) {
        lds_vg[tV * 40 + cV0 + j] = (__bf16)(xv[j] * wv_g[gc + j] + bv_g[gc + j]);
        lds_vl[tV * 40 + cV0 + j] = (__bf16)(xv[j] * wv_l[gc + j] + bv_l[gc + j]);
      }
    }
    __syncthreads();

    // --- A fragments straight from global P (rows contiguous bf16, 16B aligned)
    FragB16 ag[4], al[4];
#pragma unroll
    for (int mi = 0; mi < 4; ++mi) {
      const int row = wm * 64 + mi * 16 + ln;
      const __bf16* pg = p_ws + (((size_t)0 * 32 + bh) * 256 + row) * 256 + kc;
      const __bf16* pl = p_ws + (((size_t)1 * 32 + bh) * 256 + row) * 256 + kc;
      ag[mi].u[0] = *(const uint4*)(pg + hl * 8);
      ag[mi].u[1] = *(const uint4*)(pg + 16 + hl * 8);
      al[mi].u[0] = *(const uint4*)(pl + hl * 8);
      al[mi].u[1] = *(const uint4*)(pl + 16 + hl * 8);
    }
    FragB16 bg[2], bl[2];
#pragma unroll
    for (int ni = 0; ni < 2; ++ni) {
      const int col = wn * 32 + ni * 16 + ln;
      bg[ni].u[0] = *(const uint4*)&lds_vg[col * 40 + hl * 8];
      bg[ni].u[1] = *(const uint4*)&lds_vg[col * 40 + 16 + hl * 8];
      bl[ni].u[0] = *(const uint4*)&lds_vl[col * 40 + hl * 8];
      bl[ni].u[1] = *(const uint4*)&lds_vl[col * 40 + 16 + hl * 8];
    }
#pragma unroll
    for (int mi = 0; mi < 4; ++mi)
#pragma unroll
      for (int ni = 0; ni < 2; ++ni) {
        acc[mi][ni] = __builtin_amdgcn_wmma_f32_16x16x32_bf16(
            false, ag[mi].v, false, bg[ni].v, (short)0, acc[mi][ni],
            false, false);
        acc[mi][ni] = __builtin_amdgcn_wmma_f32_16x16x32_bf16(
            false, al[mi].v, false, bl[ni].v, (short)0, acc[mi][ni],
            false, false);
      }
    __syncthreads();
  }

  // --- stage O to LDS as [token][channel] for coalesced fp32 stores
#pragma unroll
  for (int mi = 0; mi < 4; ++mi)
#pragma unroll
    for (int ni = 0; ni < 2; ++ni)
#pragma unroll
      for (int i = 0; i < 8; ++i) {
        const int ch = wm * 64 + mi * 16 + i + 8 * hl;
        const int tk = wn * 32 + ni * 16 + ln;
        lds_o[tk * 257 + ch] = acc[mi][ni][i];
      }
  __syncthreads();

  // --- final per-channel affine + coalesced write
  {
    const int t  = tid >> 2;
    const int c0 = (tid & 3) * 64;
    float* op = out + ((size_t)b * HW_ + tbase + t) * C_ + h * DH_ + c0;
#pragma unroll 4
    for (int j = 0; j < 64; ++j) {
      const int gc = h * DH_ + c0 + j;
      op[j] = lds_o[t * 257 + c0 + j] * wp[gc] + bp[gc];
    }
  }
}

// ---------------------------------------------------------------------------
extern "C" void kernel_launch(void* const* d_in, const int* in_sizes, int n_in,
                              void* d_out, int out_size, void* d_ws, size_t ws_size,
                              hipStream_t stream) {
  const float* q    = (const float*)d_in[0];
  const float* k    = (const float*)d_in[1];
  const float* v    = (const float*)d_in[2];
  const float* wq_g = (const float*)d_in[3];
  const float* bq_g = (const float*)d_in[4];
  const float* wk_g = (const float*)d_in[5];
  const float* bk_g = (const float*)d_in[6];
  const float* wv_g = (const float*)d_in[7];
  const float* bv_g = (const float*)d_in[8];
  const float* wq_l = (const float*)d_in[9];
  const float* bq_l = (const float*)d_in[10];
  const float* wk_l = (const float*)d_in[11];
  const float* bk_l = (const float*)d_in[12];
  const float* wv_l = (const float*)d_in[13];
  const float* bv_l = (const float*)d_in[14];
  const float* wp   = (const float*)d_in[15];
  const float* bp   = (const float*)d_in[16];

  // workspace: P (bf16) [2 branches][32 bh][256][256] = 8 MiB
  const size_t ws_needed = (size_t)2 * 32 * 256 * 256 * sizeof(__bf16);
  if (ws_size < ws_needed) return;
  __bf16* p_ws = (__bf16*)d_ws;

  k_qk_softmax<<<dim3(256), dim3(256), 0, stream>>>(
      q, k, wq_g, bq_g, wk_g, bk_g, wq_l, bq_l, wk_l, bk_l, p_ws);

  k_pv_out<<<dim3(2048), dim3(256), 0, stream>>>(
      v, p_ws, wv_g, bv_g, wv_l, bv_l, wp, bp, (float*)d_out);
}
